// CausalSelfAttention_9826885173887
// MI455X (gfx1250) — matrix operand order repaired and checked by reading, not compiled, not run on hardware
//
#include <hip/hip_runtime.h>
#include <hip/hip_bf16.h>
#include <math.h>

typedef __attribute__((ext_vector_type(16))) _Float16 v16h;
typedef __attribute__((ext_vector_type(8)))  _Float16 h8;
typedef __attribute__((ext_vector_type(4)))  _Float16 h4;
typedef __attribute__((ext_vector_type(8)))  float    v8f;

// Problem constants (match reference): B=2, T=2048, C=1024, H=16, D=64
#define ATT_B   2
#define ATT_T   2048
#define ATT_C   1024
#define ATT_H   16
#define ATT_D   64

__device__ __forceinline__ v16h make_frag(h8 lo, h8 hi) {
  v16h r;
#pragma unroll
  for (int e = 0; e < 8; ++e) { r[e] = lo[e]; r[e + 8] = hi[e]; }
  return r;
}

// Async 16-byte global->LDS copy on the CDNA5 async pipe (ASYNCcnt).
// VDST = per-lane LDS byte offset (low 32 bits of the LDS-aperture address),
// VADDR = 64-bit global address, GV mode (saddr = off).
__device__ __forceinline__ void async_copy_b128(_Float16* lds_dst,
                                                const _Float16* gsrc) {
  unsigned loff = (unsigned)(size_t)lds_dst;
  asm volatile("global_load_async_to_lds_b128 %0, %1, off"
               :: "v"(loff), "v"(gsrc)
               : "memory");
}
__device__ __forceinline__ void wait_async0() {
  asm volatile("s_wait_asynccnt 0x0" ::: "memory");
}

// ---------------------------------------------------------------------------
// Kernel 1: qkv = x @ W_attn   (M=4096, N=3072, K=1024), f32 in -> f16 out,
// scattered into q/k/v buffers laid out [B, H, T, D]. b128 global loads.
// (Staging converts f32->f16 in flight, so it stays on the VGPR path.)
// ---------------------------------------------------------------------------
__global__ __launch_bounds__(256) void qkv_gemm_kernel(
    const float* __restrict__ x, const float* __restrict__ Wa,
    _Float16* __restrict__ q, _Float16* __restrict__ k, _Float16* __restrict__ v)
{
  constexpr int BM = 128, BN = 64, BK = 32, LDA = 40, LDB = 40;
  __shared__ _Float16 As[BM * LDA];      // [row][k]
  __shared__ _Float16 Bs[BN * LDB];      // transposed: [n][k]
  const int tid  = threadIdx.x;
  const int lane = tid & 31, wid = tid >> 5;
  const int h16  = lane >> 4, l16 = lane & 15;
  const int bm = blockIdx.y * BM, bn = blockIdx.x * BN;
  const int wM = (wid & 3) * 32,  wN = (wid >> 2) * 32;

  v8f acc[2][2] = {};

  for (int kk = 0; kk < ATT_C; kk += BK) {
#pragma unroll
    for (int i = 0; i < 4; ++i) {              // 128x32 A tile: 1024 float4
      int idx = i * 256 + tid;
      int r = idx >> 3, c4 = idx & 7;
      float4 t4 = *(const float4*)&x[(size_t)(bm + r) * ATT_C + kk + c4 * 4];
      h4 p; p[0] = (_Float16)t4.x; p[1] = (_Float16)t4.y;
      p[2] = (_Float16)t4.z; p[3] = (_Float16)t4.w;
      *(h4*)&As[r * LDA + c4 * 4] = p;
    }
#pragma unroll
    for (int i = 0; i < 2; ++i) {              // 32x64 B tile: 512 float4
      int idx = i * 256 + tid;
      int kr = idx >> 4, c4 = idx & 15;
      float4 t4 = *(const float4*)&Wa[(size_t)(kk + kr) * (3 * ATT_C) + bn + c4 * 4];
      Bs[(c4 * 4 + 0) * LDB + kr] = (_Float16)t4.x;
      Bs[(c4 * 4 + 1) * LDB + kr] = (_Float16)t4.y;
      Bs[(c4 * 4 + 2) * LDB + kr] = (_Float16)t4.z;
      Bs[(c4 * 4 + 3) * LDB + kr] = (_Float16)t4.w;
    }
    __syncthreads();

    v16h a[2], b[2];
#pragma unroll
    for (int i = 0; i < 2; ++i) {
      int row = wM + i * 16 + l16;
      a[i] = make_frag(*(const h8*)&As[row * LDA + h16 * 8],
                       *(const h8*)&As[row * LDA + 16 + h16 * 8]);
    }
#pragma unroll
    for (int j = 0; j < 2; ++j) {
      int n = wN + j * 16 + l16;
      b[j] = make_frag(*(const h8*)&Bs[n * LDB + h16 * 8],
                       *(const h8*)&Bs[n * LDB + 16 + h16 * 8]);
    }
#pragma unroll
    for (int i = 0; i < 2; ++i)
#pragma unroll
      for (int j = 0; j < 2; ++j)
        acc[i][j] = __builtin_amdgcn_wmma_f32_16x16x32_f16(
            false, a[i], false, b[j], (short)0, acc[i][j], false, false);
    __syncthreads();
  }

  // Epilogue: n in [bn, bn+64) lies inside one of q/k/v (bn % 64 == 0).
  const int which = bn >> 10;
  _Float16* dst = (which == 0) ? q : (which == 1) ? k : v;
  const int cbase = bn & (ATT_C - 1);
#pragma unroll
  for (int i = 0; i < 2; ++i)
#pragma unroll
    for (int j = 0; j < 2; ++j)
#pragma unroll
      for (int r = 0; r < 8; ++r) {
        int m = bm + wM + i * 16 + r + 8 * h16;
        int c = cbase + wN + j * 16 + l16;
        int hh = c >> 6, d = c & 63;
        int bb = m >> 11, t = m & (ATT_T - 1);
        dst[(((size_t)bb * ATT_H + hh) * ATT_T + t) * ATT_D + d] =
            (_Float16)acc[i][j][r];
      }
}

// ---------------------------------------------------------------------------
// Kernel 2: RoPE with YaRN scaling (T=2048 > 1024 => scale=2 branch),
// in-place on q and k. One thread per (b,h,t, i<32) rotation pair.
// ---------------------------------------------------------------------------
__global__ void rope_kernel(_Float16* __restrict__ q, _Float16* __restrict__ k, int T)
{
  int idx = blockIdx.x * blockDim.x + threadIdx.x;   // B*H*T*32 threads
  int i  = idx & 31;
  int t  = (idx >> 5) & (ATT_T - 1);
  int bh = idx >> 16;

  float fi       = (float)i * (1.0f / 32.0f);        // arange(0,64,2)/64
  float inv_freq = powf(10000.0f, -fi);
  float scale    = fmaxf(1.0f, (float)T / 1024.0f);
  float f, msc;
  if (scale > 1.0f) {
    float new_base = 10000.0f * powf(scale, 64.0f / 62.0f);
    float inv_y    = powf(new_base, -fi);
    float ramp     = fminf(fmaxf((fi - 1.0f) / 31.0f, 0.0f), 1.0f);
    f   = inv_y * (1.0f - ramp) + (inv_freq / scale) * ramp;
    msc = 0.1f * logf(scale) + 1.0f;
  } else { f = inv_freq; msc = 1.0f; }

  float ang = (float)t * f;
  float c = cosf(ang) * msc, s = sinf(ang) * msc;

  size_t base = ((size_t)bh * ATT_T + t) * ATT_D;
  float q1 = (float)q[base + i], q2 = (float)q[base + i + 32];
  q[base + i]      = (_Float16)(q1 * c - q2 * s);
  q[base + i + 32] = (_Float16)(q2 * c + q1 * s);
  float k1 = (float)k[base + i], k2 = (float)k[base + i + 32];
  k[base + i]      = (_Float16)(k1 * c - k2 * s);
  k[base + i + 32] = (_Float16)(k2 * c + k1 * s);
}

// ---------------------------------------------------------------------------
// Kernel 3: flash attention. 4 waves / block, 64 query rows per block of one
// (b,h). K tile async-copied to LDS (ASYNCcnt pipe); V staged transposed.
// Online softmax in f32. Uniform kb loop => convergent barriers; fully-masked
// tiles are exact no-ops (p == 0) once a row has real mass, which holds from
// the first tile (key kb=0 <= every query).
// ---------------------------------------------------------------------------
__global__ __launch_bounds__(128) void attn_kernel(
    const _Float16* __restrict__ q, const _Float16* __restrict__ k,
    const _Float16* __restrict__ v, _Float16* __restrict__ y)
{
  constexpr int LDK = 80;   // Ks stride (f16): [key][d], 160B rows, 16B aligned
  constexpr int LDV = 48;   // Vt stride (f16): [d][key],  96B rows, 16B aligned
  __shared__ _Float16 Ks[32 * LDK];
  __shared__ _Float16 Vt[ATT_D * LDV];
  __shared__ _Float16 Pt[4 * 16 * 32];
  const int tid  = threadIdx.x;
  const int lane = tid & 31, wid = tid >> 5;
  const int h16  = lane >> 4, l16 = lane & 15;
  const int bh     = blockIdx.y;             // b*H + h
  const int qbase0 = blockIdx.x * 64;        // block's 64 query rows
  const int qbase  = qbase0 + wid * 16;      // this wave's 16 query rows
  const size_t hb  = (size_t)bh * ATT_T * ATT_D;
  _Float16* Pw = &Pt[wid * 16 * 32];

  // Q fragments (two d-halves), rows = qbase..qbase+15
  v16h aq[2];
#pragma unroll
  for (int c = 0; c < 2; ++c) {
    const _Float16* qp = q + hb + (size_t)(qbase + l16) * ATT_D + c * 32 + h16 * 8;
    aq[c] = make_frag(*(const h8*)qp, *(const h8*)(qp + 16));
  }

  v8f acc[4] = {};
  float mrow[8], lrow[8];
#pragma unroll
  for (int r = 0; r < 8; ++r) { mrow[r] = -3.0e38f; lrow[r] = 0.0f; }

  for (int kb = 0; kb <= qbase0 + 63; kb += 32) {
    // ---- stage K tile (async b128 to LDS) + transposed V tile ------------
#pragma unroll
    for (int i = 0; i < 2; ++i) {
      int idx = i * 128 + tid;               // 256 chunks of 8 f16
      int key = idx >> 3, c8 = idx & 7;
      async_copy_b128(&Ks[key * LDK + c8 * 8],
                      &k[hb + (size_t)(kb + key) * ATT_D + c8 * 8]);
      h8 vv = *(const h8*)&v[hb + (size_t)(kb + key) * ATT_D + c8 * 8];
#pragma unroll
      for (int e = 0; e < 8; ++e) Vt[(c8 * 8 + e) * LDV + key] = vv[e];
    }
    wait_async0();           // drain this wave's async copies
    __syncthreads();         // publish Ks/Vt across all 4 waves

    // ---- S = (Q . K^T) / sqrt(D), causal-masked --------------------------
    float sv[2][8];
#pragma unroll
    for (int s = 0; s < 2; ++s) {
      v8f cfr = {};
#pragma unroll
      for (int c = 0; c < 2; ++c) {
        const _Float16* kp = &Ks[(s * 16 + l16) * LDK + c * 32 + h16 * 8];
        v16h bk = make_frag(*(const h8*)kp, *(const h8*)(kp + 16));
        cfr = __builtin_amdgcn_wmma_f32_16x16x32_f16(
            false, aq[c], false, bk, (short)0, cfr, false, false);
      }
#pragma unroll
      for (int r = 0; r < 8; ++r) {
        int qrow = qbase + r + 8 * h16;
        int key  = kb + s * 16 + l16;
        float val = cfr[r] * 0.125f;         // 1/sqrt(64)
        sv[s][r] = (key > qrow) ? -3.0e38f : val;
      }
    }
    // ---- online softmax (row reductions across 16-lane halves) -----------
    float pv[2][8];
#pragma unroll
    for (int r = 0; r < 8; ++r) {
      float tmax = fmaxf(sv[0][r], sv[1][r]);
#pragma unroll
      for (int off = 1; off < 16; off <<= 1)
        tmax = fmaxf(tmax, __shfl_xor(tmax, off, 32));
      float mnew  = fmaxf(mrow[r], tmax);
      float alpha = __expf(mrow[r] - mnew);
      float p0 = __expf(sv[0][r] - mnew);
      float p1 = __expf(sv[1][r] - mnew);
      float rsum = p0 + p1;
#pragma unroll
      for (int off = 1; off < 16; off <<= 1)
        rsum += __shfl_xor(rsum, off, 32);
      lrow[r] = lrow[r] * alpha + rsum;
      mrow[r] = mnew;
#pragma unroll
      for (int j = 0; j < 4; ++j) acc[j][r] *= alpha;
      pv[0][r] = p0; pv[1][r] = p1;
    }
    // ---- P: C-layout -> A-fragment via per-wave LDS bounce ---------------
#pragma unroll
    for (int r = 0; r < 8; ++r) {
      int mloc = r + 8 * h16;
      Pw[mloc * 32 + l16]      = (_Float16)pv[0][r];
      Pw[mloc * 32 + 16 + l16] = (_Float16)pv[1][r];
    }
    __syncthreads();
    v16h ap = make_frag(*(const h8*)&Pw[l16 * 32 + h16 * 8],
                        *(const h8*)&Pw[l16 * 32 + 16 + h16 * 8]);
    // ---- acc += P . V (V read transposed from LDS) -----------------------
#pragma unroll
    for (int j = 0; j < 4; ++j) {
      const _Float16* vp = &Vt[(j * 16 + l16) * LDV + h16 * 8];
      v16h bv = make_frag(*(const h8*)vp, *(const h8*)(vp + 16));
      acc[j] = __builtin_amdgcn_wmma_f32_16x16x32_f16(
          false, ap, false, bv, (short)0, acc[j], false, false);
    }
    __syncthreads();   // all waves done with Ks/Vt/P before restaging
  }

#pragma unroll
  for (int r = 0; r < 8; ++r) {
    float inv = 1.0f / lrow[r];
    int row = qbase + r + 8 * h16;
#pragma unroll
    for (int j = 0; j < 4; ++j)
      y[hb + (size_t)row * ATT_D + j * 16 + l16] = (_Float16)(acc[j][r] * inv);
  }
}

// ---------------------------------------------------------------------------
// Kernel 4: out = y @ W_proj  (M=4096, N=1024, K=1024). A tile (f16 y,
// verbatim copy) goes through the async global->LDS pipe; f32 output.
// ---------------------------------------------------------------------------
__global__ __launch_bounds__(256) void proj_gemm_kernel(
    const _Float16* __restrict__ y, const float* __restrict__ Wp,
    float* __restrict__ out)
{
  constexpr int BM = 128, BN = 64, BK = 32, LDA = 40, LDB = 40;
  __shared__ _Float16 As[BM * LDA];
  __shared__ _Float16 Bs[BN * LDB];
  const int tid  = threadIdx.x;
  const int lane = tid & 31, wid = tid >> 5;
  const int h16  = lane >> 4, l16 = lane & 15;
  const int bm = blockIdx.y * BM, bn = blockIdx.x * BN;
  const int wM = (wid & 3) * 32,  wN = (wid >> 2) * 32;

  v8f acc[2][2] = {};

  for (int kk = 0; kk < ATT_C; kk += BK) {
#pragma unroll
    for (int i = 0; i < 2; ++i) {            // 128x32 f16 A tile: async b128
      int idx = i * 256 + tid;
      int r = idx >> 2, c8 = idx & 3;        // 4 chunks of 8 per row
      int m = bm + r, kc = kk + c8 * 8;      // chunk stays inside one head
      int bb = m >> 11, t = m & (ATT_T - 1), hh = kc >> 6, d = kc & 63;
      async_copy_b128(&As[r * LDA + c8 * 8],
                      &y[(((size_t)bb * ATT_H + hh) * ATT_T + t) * ATT_D + d]);
    }
#pragma unroll
    for (int i = 0; i < 2; ++i) {            // 32x64 f32 B tile: 512 float4
      int idx = i * 256 + tid;
      int kr = idx >> 4, c4 = idx & 15;
      float4 t4 = *(const float4*)&Wp[(size_t)(kk + kr) * ATT_C + bn + c4 * 4];
      Bs[(c4 * 4 + 0) * LDB + kr] = (_Float16)t4.x;
      Bs[(c4 * 4 + 1) * LDB + kr] = (_Float16)t4.y;
      Bs[(c4 * 4 + 2) * LDB + kr] = (_Float16)t4.z;
      Bs[(c4 * 4 + 3) * LDB + kr] = (_Float16)t4.w;
    }
    wait_async0();
    __syncthreads();

    v16h a[2], b[2];
#pragma unroll
    for (int i = 0; i < 2; ++i) {
      int row = wM + i * 16 + l16;
      a[i] = make_frag(*(const h8*)&As[row * LDA + h16 * 8],
                       *(const h8*)&As[row * LDA + 16 + h16 * 8]);
    }
#pragma unroll
    for (int j = 0; j < 2; ++j) {
      int n = wN + j * 16 + l16;
      b[j] = make_frag(*(const h8*)&Bs[n * LDB + h16 * 8],
                       *(const h8*)&Bs[n * LDB + 16 + h16 * 8]);
    }
#pragma unroll
    for (int i = 0; i < 2; ++i)
#pragma unroll
      for (int j = 0; j < 2; ++j)
        acc[i][j] = __builtin_amdgcn_wmma_f32_16x16x32_f16(
            false, a[i], false, b[j], (short)0, acc[i][j], false, false);
    __syncthreads();
  }

#pragma unroll
  for (int i = 0; i < 2; ++i)
#pragma unroll
    for (int j = 0; j < 2; ++j)
#pragma unroll
      for (int r = 0; r < 8; ++r) {
        int m = bm + wM + i * 16 + r + 8 * h16;
        int n = bn + wN + j * 16 + l16;
        out[(size_t)m * ATT_C + n] = acc[i][j][r];
      }
}

// ---------------------------------------------------------------------------
extern "C" void kernel_launch(void* const* d_in, const int* in_sizes, int n_in,
                              void* d_out, int out_size, void* d_ws, size_t ws_size,
                              hipStream_t stream) {
  const float* x  = (const float*)d_in[0];   // [2, 2048, 1024] f32
  const float* Wa = (const float*)d_in[1];   // [1024, 3072] f32
  const float* Wp = (const float*)d_in[2];   // [1024, 1024] f32
  float* out = (float*)d_out;                // [2, 2048, 1024] f32

  const size_t NE = (size_t)ATT_B * ATT_H * ATT_T * ATT_D;  // 4,194,304
  char* wsb = (char*)d_ws;
  _Float16* q = (_Float16*)(wsb);
  _Float16* k = (_Float16*)(wsb + 2 * NE);
  _Float16* v = (_Float16*)(wsb + 4 * NE);
  _Float16* y = (_Float16*)(wsb + 6 * NE);

  // 1) QKV projection: (4096 x 1024) @ (1024 x 3072)
  qkv_gemm_kernel<<<dim3(3 * ATT_C / 64, ATT_B * ATT_T / 128), 256, 0, stream>>>(
      x, Wa, q, k, v);
  // 2) RoPE/YaRN on q,k
  int rope_threads = ATT_B * ATT_H * ATT_T * 32;
  rope_kernel<<<rope_threads / 256, 256, 0, stream>>>(q, k, ATT_T);
  // 3) Flash attention: 4 waves / 64 queries per block, shared K/V tiles
  attn_kernel<<<dim3(ATT_T / 64, ATT_B * ATT_H), 128, 0, stream>>>(q, k, v, y);
  // 4) Output projection: (4096 x 1024) @ (1024 x 1024)
  proj_gemm_kernel<<<dim3(ATT_C / 64, ATT_B * ATT_T / 128), 256, 0, stream>>>(
      y, Wp, out);
}
